// RNNetwork_44805098832002
// MI455X (gfx1250) — compile-verified, matrix operand order checked by reading, not run on hardware
//
#include <hip/hip_runtime.h>
#include <hip/hip_bf16.h>
#include <math.h>

// ---------------------------------------------------------------------------
// Types for CDNA5 WMMA
// ---------------------------------------------------------------------------
typedef __attribute__((ext_vector_type(16))) __bf16 v16bf;
typedef __attribute__((ext_vector_type(8)))  float  v8f;

union FragU { unsigned u[8]; v16bf v; };

// ---------------------------------------------------------------------------
// Model constants
// ---------------------------------------------------------------------------
#define NPART      8192
#define TSTEPS     1024
#define E_CRIT_NEW 26267.160775850585f
#define G_STD      0.01f
#define F_STD      0.005f
#define LOG_NU     3.6862316527834187f   // log(1/(G_STD*sqrt(2*pi)))
#define LOG_N      9.0109133472596f      // log(8192)

#define ROWS_WG    32                    // particles per workgroup
#define APITCH     516                   // A-tile pitch (dwords): 512 + 4 pad
#define BCHUNK_DW  (512 * 16)            // one B chunk: 512 cols x 32 K bf16 (dwords)

// CDNA5 async global->LDS path (guarded; falls back to sync staging)
#if defined(__has_builtin)
#  if __has_builtin(__builtin_amdgcn_global_load_async_to_lds_b128) && \
      __has_builtin(__builtin_amdgcn_s_wait_asynccnt)
#    define USE_ASYNC_LDS 1
#  endif
#endif
#ifndef USE_ASYNC_LDS
#  define USE_ASYNC_LDS 0
#endif

#define AS_GLOBAL __attribute__((address_space(1)))
#define AS_LDS    __attribute__((address_space(3)))

// The async builtin takes pointers to 16B int vectors in AS1 (global) / AS3 (LDS).
typedef int v4i_vs __attribute__((vector_size(16)));
typedef AS_GLOBAL v4i_vs* g_v4i_p;
typedef AS_LDS    v4i_vs* l_v4i_p;

// ---------------------------------------------------------------------------
// Helpers
// ---------------------------------------------------------------------------
__device__ __forceinline__ unsigned pcg_hash(unsigned x) {
    x = x * 747796405u + 2891336453u;
    unsigned w = ((x >> ((x >> 28u) + 4u)) ^ x) * 277803737u;
    return (w >> 22u) ^ w;
}

__device__ __forceinline__ float u01(unsigned h) {
    return (float)(h >> 8) * (1.0f / 16777216.0f);
}

__device__ __forceinline__ float sigmoidf(float x) {
    return 1.0f / (1.0f + __expf(-x));
}

__device__ __forceinline__ float voc_f(float s) {
    const float v_L = -1.59614486f, v_0 = 4.13646328f;
    const float gamma = 0.63726463f, alpha = 1.40174122f, beta = 2.54478965f;
    const float exp_neg_beta = 0.07848646f; // exp(-2.54478965)
    return v_L + (v_0 - v_L) * __expf(gamma * (s - 1.0f))
         + alpha * v_L * (s - 1.0f)
         + (1.0f - alpha) * v_L * (exp_neg_beta - __expf(-beta * sqrtf(fmaxf(s, 0.0f))));
}

// 16x32 bf16 A-fragment from LDS (row-major, APITCH dwords/row).
// CDNA5 16-bit A layout: lane m=l&15, half h=l>>4; VGPR v<4 -> K pair (v+4h),
// v>=4 -> K pair (v+4+4h).
__device__ __forceinline__ v16bf load_a_frag(const unsigned* ldsA, int kk, int lane) {
    const int m = lane & 15, h = lane >> 4;
    const unsigned* row = ldsA + m * APITCH + kk * 16;
    FragU f;
#pragma unroll
    for (int v = 0; v < 8; ++v) {
        int idx = v + 4 * h + ((v >= 4) ? 4 : 0);
        f.u[v] = row[idx];
    }
    return f.v;
}

// B chunk LDS layout: col c occupies 16 dwords at c*16, with group-of-4-dword
// XOR swizzle pg = g ^ (c&3) ^ ((c>>2)&3)  -> bank-conflict-free frag reads,
// 16B alignment preserved for b128 async writes/reads.
__device__ __forceinline__ v16bf load_b_frag_lds(const unsigned* ldsB, int j0, int lane) {
    const int c = j0 + (lane & 15);
    const int h = lane >> 4;
    const int sw = (c & 3) ^ ((c >> 2) & 3);
    const unsigned* colp = ldsB + c * 16;
    const unsigned* p0 = colp + 4 * ((h)     ^ sw);  // idx group h     -> f.u[0..3]
    const unsigned* p1 = colp + 4 * ((h + 2) ^ sw);  // idx group h + 2 -> f.u[4..7]
    FragU f;
#pragma unroll
    for (int q = 0; q < 4; ++q) { f.u[q] = p0[q]; f.u[4 + q] = p1[q]; }
    return f.v;
}

// Stage one B chunk (512 cols x 32 K bf16 = 32KB) into LDS; 2048 b128 moves,
// 8 per thread. Async on CDNA5 (ASYNCcnt), sync copy otherwise.
__device__ __forceinline__ void stage_b_chunk(const unsigned* __restrict__ w2d,
                                              unsigned* ldsB, int kk, int tid) {
#pragma unroll
    for (int i = 0; i < 8; ++i) {
        const int e = tid * 8 + i;        // 0..2047
        const int c = e >> 2;             // column 0..511
        const int g = e & 3;              // dword group 0..3
        const int pg = g ^ (c & 3) ^ ((c >> 2) & 3);
        const unsigned* src = w2d + c * 512 + kk * 16 + 4 * g;
        unsigned* dst = ldsB + c * 16 + 4 * pg;
#if USE_ASYNC_LDS
        __builtin_amdgcn_global_load_async_to_lds_b128(
            (g_v4i_p)(const void*)src, (l_v4i_p)(void*)dst, 0, 0);
#else
        *(uint4*)dst = *(const uint4*)src;
#endif
    }
}

// ---------------------------------------------------------------------------
// Prep: convert W2 fp32 -> bf16, zero loss accumulator
// ---------------------------------------------------------------------------
__global__ __launch_bounds__(256) void prep_kernel(const float* __restrict__ W2,
                                                   __bf16* __restrict__ w2bf,
                                                   float* __restrict__ loss_ws) {
    int i = blockIdx.x * 256 + threadIdx.x;
    if (i < 512 * 1024) w2bf[i] = (__bf16)W2[i];
    if (i == 0) loss_ws[0] = 0.0f;
}

// ---------------------------------------------------------------------------
// Per-step fused kernel: SOC propagate + layer1 + WMMA layer2 + layer3 + V,logW
// One block per 32 particles. 256 threads = 8 wave32.
// t == -1 : init pass (V0 from soc_init, no history / no logW).
// ---------------------------------------------------------------------------
__global__ __launch_bounds__(256) void step_kernel(
    int t,
    const float* __restrict__ soc_init,
    float* __restrict__ soc_carry,
    float* __restrict__ soc_cur,
    const unsigned* __restrict__ w2d,      // bf16 W2 viewed as dwords
    const float* __restrict__ W1,          // (1024,2)
    const float* __restrict__ b1,          // (1024,)
    const float* __restrict__ b2,          // (512,)
    const float* __restrict__ W3,          // (512,)
    const float* __restrict__ b3,          // (1,)
    const float* __restrict__ current,     // (1024,)
    const float* __restrict__ vmeas,       // (1024,)
    const float* __restrict__ E_crit,      // (1,)
    float* __restrict__ Vbuf,
    float* __restrict__ logW,
    float* __restrict__ outf)              // d_out as float
{
    __shared__ unsigned ldsA[ROWS_WG * APITCH];   // 32 x 1024 bf16 (padded) ~66KB
    __shared__ unsigned ldsB[2 * BCHUNK_DW];      // double-buffered B chunks 64KB
    __shared__ float socv[ROWS_WG];
    __shared__ float rowsum[ROWS_WG];

    const int tid   = threadIdx.x;
    const int nbase = blockIdx.x * ROWS_WG;

    // kick off async staging of the first two B chunks; overlaps with phases 0-1
    stage_b_chunk(w2d, ldsB,             0, tid);
    stage_b_chunk(w2d, ldsB + BCHUNK_DW, 1, tid);

    // ---- phase 0: propagate SOC (32 lanes) ----
    if (tid < ROWS_WG) {
        const int n = nbase + tid;
        float s;
        if (t < 0) {
            s = soc_init[n];
            soc_carry[n] = s;
        } else {
            const float Iprev = current[(t == 0) ? 0 : (t - 1)];
            s = soc_carry[n] - Iprev * Vbuf[n] * (E_crit[0] / E_CRIT_NEW);
            unsigned ha = pcg_hash((unsigned)n * 1664525u + (unsigned)(t + 1) * 1013904223u);
            unsigned hb = pcg_hash(ha ^ 0x85EBCA6Bu);
            float uu1 = u01(ha) + 5.96e-8f;
            float uu2 = u01(hb);
            float g = sqrtf(-2.0f * __logf(uu1)) * __cosf(6.28318530718f * uu2);
            s = s + F_STD * g;
            s = (s > 1.0f) ? 1.0f : ((s < 0.0f) ? 1e-10f : s);
        }
        socv[tid]   = s;
        soc_cur[n]  = s;
        rowsum[tid] = 0.0f;
    }
    __syncthreads();

    // ---- phase 1: layer 1 (2 -> 1024, sigmoid) straight into LDS as bf16 ----
    const float It = current[(t < 0) ? 0 : t];
    const float sI = (It + 2.0f) * 0.25f;   // (I - C_MIN)/(C_MAX - C_MIN)

    float w1a[4], cc[4];
#pragma unroll
    for (int j = 0; j < 4; ++j) {
        const int k = tid + 256 * j;
        w1a[j] = W1[2 * k];
        cc[j]  = sI * W1[2 * k + 1] + b1[k];
    }
    __bf16* ldsH = (__bf16*)ldsA;
#pragma unroll 4
    for (int r = 0; r < ROWS_WG; ++r) {
        const float s = socv[r];
#pragma unroll
        for (int j = 0; j < 4; ++j) {
            const float h = sigmoidf(s * w1a[j] + cc[j]);
            ldsH[r * (2 * APITCH) + tid + 256 * j] = (__bf16)h;
        }
    }
    // no barrier here: the wait+barrier at the top of the K-loop covers phase 1

    // ---- phase 2: layer 2 GEMM via v_wmma_f32_16x16x32_bf16 ----
    // wave w owns output columns [64w, 64w+64): 4 col-tiles x 2 row-tiles.
    const int wave = tid >> 5;
    const int lane = tid & 31;

    v8f acc[4][2];
#pragma unroll
    for (int ct = 0; ct < 4; ++ct) {
        acc[ct][0] = (v8f){0, 0, 0, 0, 0, 0, 0, 0};
        acc[ct][1] = (v8f){0, 0, 0, 0, 0, 0, 0, 0};
    }

    for (int kk = 0; kk < 32; ++kk) {
#if USE_ASYNC_LDS
        if (kk < 31) __builtin_amdgcn_s_wait_asynccnt(8);   // chunk kk landed
        else         __builtin_amdgcn_s_wait_asynccnt(0);
#endif
        __syncthreads();   // chunk kk (and, first time, the A tile) visible

        const v16bf a0 = load_a_frag(ldsA,                kk, lane);
        const v16bf a1 = load_a_frag(ldsA + 16 * APITCH,  kk, lane);
        const unsigned* bbuf = ldsB + (kk & 1) * BCHUNK_DW;
#pragma unroll
        for (int ct = 0; ct < 4; ++ct) {
            const v16bf bfrag = load_b_frag_lds(bbuf, wave * 64 + ct * 16, lane);
            acc[ct][0] = __builtin_amdgcn_wmma_f32_16x16x32_bf16(
                false, a0, false, bfrag, (short)0, acc[ct][0], false, false);
            acc[ct][1] = __builtin_amdgcn_wmma_f32_16x16x32_bf16(
                false, a1, false, bfrag, (short)0, acc[ct][1], false, false);
        }
        __syncthreads();   // all waves done reading buffer (kk&1)
        if (kk < 30) stage_b_chunk(w2d, ldsB + (kk & 1) * BCHUNK_DW, kk + 2, tid);
    }

    // ---- phase 3: bias + sigmoid + layer-3 partial dot (ds_add_f32) ----
#pragma unroll
    for (int ct = 0; ct < 4; ++ct) {
        const int col = wave * 64 + ct * 16 + (lane & 15);
        const float w3c = W3[col];
        const float b2c = b2[col];
#pragma unroll
        for (int rt = 0; rt < 2; ++rt) {
#pragma unroll
            for (int r = 0; r < 8; ++r) {
                const float h2 = sigmoidf(acc[ct][rt][r] + b2c);
                const int row = rt * 16 + ((lane < 16) ? r : (r + 8));
                atomicAdd(&rowsum[row], h2 * w3c);
            }
        }
    }
    __syncthreads();

    // ---- phase 4: finalize V, logW, history ----
    if (tid < ROWS_WG) {
        const int n = nbase + tid;
        const float Z = rowsum[tid] + b3[0];
        const float s = socv[tid];
        const float V = voc_f(s) - It * Z;
        Vbuf[n] = V;
        if (t >= 0) {
            const float d = (V - vmeas[t]) * (1.0f / G_STD);
            logW[n] = LOG_NU - 0.5f * d * d;
            outf[1 + (size_t)n * TSTEPS + t] = V;   // v_hist.T layout (N,T)
        }
    }
}

// ---------------------------------------------------------------------------
// Per-step tail: logsumexp loss, cumsum, systematic resample, s_hist
// Single block of 1024 threads; 8 particles per thread.
// ---------------------------------------------------------------------------
__global__ __launch_bounds__(1024) void resample_kernel(
    int t,
    const float* __restrict__ logW,
    const float* __restrict__ soc_cur,
    float* __restrict__ soc_carry,
    float* __restrict__ loss_ws,
    float* __restrict__ outf)
{
    __shared__ float cw[NPART];   // 32KB cumulative weights
    __shared__ float red[1024];

    const int tid = threadIdx.x;

    float lw[8];
    float mx = -3.4e38f;
#pragma unroll
    for (int j = 0; j < 8; ++j) {
        lw[j] = logW[tid * 8 + j];
        mx = fmaxf(mx, lw[j]);
    }
    red[tid] = mx;
    __syncthreads();
    for (int s = 512; s > 0; s >>= 1) {
        if (tid < s) red[tid] = fmaxf(red[tid], red[tid + s]);
        __syncthreads();
    }
    const float maxv = red[0];
    __syncthreads();

    float c[8];
    float csum = 0.0f;
#pragma unroll
    for (int j = 0; j < 8; ++j) {
        csum += __expf(lw[j] - maxv);
        c[j] = csum;
    }
    red[tid] = csum;
    __syncthreads();

    for (int off = 1; off < 1024; off <<= 1) {
        float add = (tid >= off) ? red[tid - off] : 0.0f;
        __syncthreads();
        red[tid] += add;
        __syncthreads();
    }
    const float total  = red[1023];
    const float offset = red[tid] - csum;  // exclusive prefix
#pragma unroll
    for (int j = 0; j < 8; ++j) cw[tid * 8 + j] = offset + c[j];
    __syncthreads();

    if (tid == 0) {
        const float loss = loss_ws[0] + maxv + __logf(total) - LOG_N;
        loss_ws[0] = loss;
        if (t == TSTEPS - 1) outf[0] = loss;
    }

    const unsigned hu = pcg_hash(0x000A5117u ^ ((unsigned)(t + 1) * 0x9E3779B9u));
    const float u0 = u01(hu);

#pragma unroll
    for (int j = 0; j < 8; ++j) {
        const int n = tid * 8 + j;
        const float target = (u0 + (float)n) * (1.0f / (float)NPART) * total;
        int lo = 0, hi = NPART;
        while (lo < hi) {                      // searchsorted, side='left'
            const int mid = (lo + hi) >> 1;
            if (cw[mid] < target) lo = mid + 1; else hi = mid;
        }
        const int idx = (lo > NPART - 1) ? (NPART - 1) : lo;
        const float s = soc_cur[idx];
        soc_carry[n] = s;
        outf[1 + (size_t)NPART * TSTEPS + (size_t)n * TSTEPS + t] = s;  // s_hist.T
    }
}

// ---------------------------------------------------------------------------
// Host entry
// ---------------------------------------------------------------------------
extern "C" void kernel_launch(void* const* d_in, const int* in_sizes, int n_in,
                              void* d_out, int out_size, void* d_ws, size_t ws_size,
                              hipStream_t stream) {
    (void)in_sizes; (void)n_in; (void)out_size; (void)ws_size;

    const float* soc_init = (const float*)d_in[0];
    const float* current  = (const float*)d_in[1];
    const float* vmeas    = (const float*)d_in[2];
    const float* W1       = (const float*)d_in[3];
    const float* b1       = (const float*)d_in[4];
    const float* W2       = (const float*)d_in[5];
    const float* b2       = (const float*)d_in[6];
    const float* W3       = (const float*)d_in[7];
    const float* b3       = (const float*)d_in[8];
    const float* E_crit   = (const float*)d_in[9];

    float* outf = (float*)d_out;

    char* ws = (char*)d_ws;
    __bf16*   w2bf      = (__bf16*)ws;                         // 1 MB
    unsigned* w2d       = (unsigned*)ws;
    float*    Vbuf      = (float*)(ws + (1 << 20));            // 32 KB
    float*    logWb     = (float*)(ws + (1 << 20) + 32768);    // 32 KB
    float*    soc_carry = (float*)(ws + (1 << 20) + 65536);    // 32 KB
    float*    soc_cur   = (float*)(ws + (1 << 20) + 98304);    // 32 KB
    float*    loss_ws   = (float*)(ws + (1 << 20) + 131072);   // 4 B

    prep_kernel<<<2048, 256, 0, stream>>>(W2, w2bf, loss_ws);

    // init pass: V0 from soc_init (t = -1)
    step_kernel<<<NPART / ROWS_WG, 256, 0, stream>>>(
        -1, soc_init, soc_carry, soc_cur, w2d, W1, b1, b2, W3, b3,
        current, vmeas, E_crit, Vbuf, logWb, outf);

    for (int t = 0; t < TSTEPS; ++t) {
        step_kernel<<<NPART / ROWS_WG, 256, 0, stream>>>(
            t, soc_init, soc_carry, soc_cur, w2d, W1, b1, b2, W3, b3,
            current, vmeas, E_crit, Vbuf, logWb, outf);
        resample_kernel<<<1, 1024, 0, stream>>>(
            t, logWb, soc_cur, soc_carry, loss_ws, outf);
    }
}
